// WindowTimeAttention_81475529605588
// MI455X (gfx1250) — matrix operand order repaired and checked by reading, not compiled, not run on hardware
//
#include <hip/hip_runtime.h>
#include <hip/hip_bf16.h>

typedef __attribute__((ext_vector_type(16))) __bf16 bf16x16;
typedef __attribute__((ext_vector_type(8)))  __bf16 bf16x8;
typedef __attribute__((ext_vector_type(8)))  float  f32x8;

// Problem constants
static constexpr int    B_   = 1024;
static constexpr int    T_   = 8;
static constexpr int    N_   = 49;
static constexpr int    C_   = 96;
static constexpr int    H_   = 3;     // heads
static constexpr int    HD_  = 32;    // head dim
static constexpr int    NW_  = 64;    // windows
static constexpr size_t TOK  = (size_t)B_ * T_ * N_;      // 401408 tokens
static constexpr float  SCALE = 0.17677669529663689f;     // 32^-0.5

__device__ __forceinline__ f32x8 wmma_bf16(bf16x16 a, bf16x16 b, f32x8 c) {
  return __builtin_amdgcn_wmma_f32_16x16x32_bf16(false, a, false, b, (short)0, c,
                                                 false, false);
}

__device__ __forceinline__ bf16x16 ld16bf(const __bf16* p) {
  const bf16x8* q = (const bf16x8*)p;
  bf16x8 lo = q[0], hi = q[1];
  bf16x16 r;
#pragma unroll
  for (int i = 0; i < 8; ++i) { r[i] = lo[i]; r[i + 8] = hi[i]; }
  return r;
}

__device__ __forceinline__ bf16x16 zero16bf() {
  bf16x16 r;
#pragma unroll
  for (int i = 0; i < 16; ++i) r[i] = (__bf16)0.0f;
  return r;
}

__device__ __forceinline__ f32x8 zero8f() {
  f32x8 r;
#pragma unroll
  for (int i = 0; i < 8; ++i) r[i] = 0.0f;
  return r;
}

// ---------------------------------------------------------------------------
// Kernel 1: QKV GEMM.  (TOK x 96) f32  @  (96 x 288) f32  + bias -> bf16 out.
// 256 threads = 8 waves; each wave owns one 16-row M tile, loops all 18 N tiles.
// ---------------------------------------------------------------------------
__global__ void __launch_bounds__(256)
qkv_gemm_kernel(const float* __restrict__ x, const float* __restrict__ w,
                const float* __restrict__ bias, __bf16* __restrict__ out) {
  __shared__ __bf16 wT[288 * 104];          // w transposed, bf16, padded stride
  for (int i = threadIdx.x; i < 288 * 96; i += 256) {
    int n = i / 96, k = i % 96;
    wT[n * 104 + k] = (__bf16)w[(size_t)k * 288 + n];
  }
  __syncthreads();

  const int lane = threadIdx.x & 31;
  const int wv   = threadIdx.x >> 5;
  const int m0   = (blockIdx.x * 8 + wv) * 16;          // TOK % 16 == 0
  const int hh   = lane >> 4;                           // lane half (0/1)
  const int ln   = lane & 15;

  // A tiles: 3 K-chunks of 32, contiguous k order (element i = k 16*hh + i), same as B.
  bf16x16 a[3];
#pragma unroll
  for (int kc = 0; kc < 3; ++kc) {
    const float* xr = x + (size_t)(m0 + ln) * 96 + kc * 32 + hh * 16;
#pragma unroll
    for (int i = 0; i < 8; ++i) {
      a[kc][i]     = (__bf16)xr[i];
      a[kc][i + 8] = (__bf16)xr[8 + i];
    }
  }

  for (int nt = 0; nt < 18; ++nt) {
    const float bv = bias[nt * 16 + ln];
    f32x8 c;
#pragma unroll
    for (int r = 0; r < 8; ++r) c[r] = bv;
#pragma unroll
    for (int kc = 0; kc < 3; ++kc) {
      bf16x16 bm = ld16bf(&wT[(nt * 16 + ln) * 104 + kc * 32 + hh * 16]);
      c = wmma_bf16(a[kc], bm, c);
    }
#pragma unroll
    for (int r = 0; r < 8; ++r) {
      out[(size_t)(m0 + r + hh * 8) * 288 + nt * 16 + ln] = (__bf16)c[r];
    }
  }
}

// ---------------------------------------------------------------------------
// Kernel 2: spatial window attention. One block per sequence (8192), 3 waves =
// 3 heads. L=49 padded to 64. K=hd=32 -> one WMMA per 16x16 score tile.
// ---------------------------------------------------------------------------
__global__ void __launch_bounds__(96)
spatial_attn_kernel(const __bf16* __restrict__ qkv, const float* __restrict__ mask,
                    __bf16* __restrict__ attn_out) {
  __shared__ __bf16 vT[3][32 * 72];   // v transposed per head: vT[d][pos]
  __shared__ __bf16 Pm[3][16 * 72];   // probability tile staging (row-major)

  const int lane = threadIdx.x & 31;
  const int h    = threadIdx.x >> 5;           // head
  const int s    = blockIdx.x;                 // sequence (b*T+t)
  const int wdx  = s & (NW_ - 1);              // window = s % 64
  const int hh   = lane >> 4;
  const int ln   = lane & 15;
  const size_t base = (size_t)s * 49 * 288;

  // Stage V transposed into LDS (pad pos 49..63 with zeros).
  {
    const int d = lane;                        // 0..31
    for (int pos = 0; pos < 64; ++pos) {
      __bf16 v = (__bf16)0.0f;
      if (pos < 49) v = qkv[base + (size_t)pos * 288 + 192 + h * 32 + d];
      vT[h][d * 72 + pos] = v;
    }
  }

  // B operand for Q@K^T: 4 N tiles over key positions; element k of GEMM = d.
  bf16x16 bmat[4];
#pragma unroll
  for (int nt = 0; nt < 4; ++nt) {
    const int posn = nt * 16 + ln;
    if (posn < 49)
      bmat[nt] = ld16bf(qkv + base + (size_t)posn * 288 + 96 + h * 32 + hh * 16);
    else
      bmat[nt] = zero16bf();
  }

  for (int mt = 0; mt < 4; ++mt) {
    // A operand: query rows mt*16 .. mt*16+15, K chunk = full hd (32).
    const int rowp  = mt * 16 + ln;
    const bool rv   = rowp < 49;
    const __bf16* qp = qkv + base + (size_t)rowp * 288 + h * 32;
    const int o1 = hh * 16;
    bf16x16 aq;
#pragma unroll
    for (int i = 0; i < 8; ++i) {
      aq[i]     = rv ? qp[o1 + i]      : (__bf16)0.0f;
      aq[i + 8] = rv ? qp[8 + o1 + i] : (__bf16)0.0f;
    }

    // Scores
    f32x8 sacc[4];
#pragma unroll
    for (int nt = 0; nt < 4; ++nt) sacc[nt] = wmma_bf16(aq, bmat[nt], zero8f());

    // scale + window mask + pad columns with -inf
#pragma unroll
    for (int nt = 0; nt < 4; ++nt) {
      const int n = nt * 16 + ln;
#pragma unroll
      for (int r = 0; r < 8; ++r) {
        const int m = mt * 16 + r + hh * 8;
        float v = sacc[nt][r] * SCALE;
        if (n >= 49)       v = -1e30f;
        else if (m < 49)   v += mask[(size_t)wdx * 2401 + m * 49 + n];
        sacc[nt][r] = v;
      }
    }

    // row softmax: reduce over 4 nt values + 16 lanes of the half-wave
#pragma unroll
    for (int r = 0; r < 8; ++r) {
      float mx = sacc[0][r];
#pragma unroll
      for (int nt = 1; nt < 4; ++nt) mx = fmaxf(mx, sacc[nt][r]);
#pragma unroll
      for (int off = 1; off < 16; off <<= 1)
        mx = fmaxf(mx, __shfl_xor(mx, off, 32));
      float ssum = 0.0f;
#pragma unroll
      for (int nt = 0; nt < 4; ++nt) {
        float e = __expf(sacc[nt][r] - mx);
        sacc[nt][r] = e;
        ssum += e;
      }
#pragma unroll
      for (int off = 1; off < 16; off <<= 1)
        ssum += __shfl_xor(ssum, off, 32);
      const float inv = 1.0f / ssum;
#pragma unroll
      for (int nt = 0; nt < 4; ++nt) sacc[nt][r] *= inv;
    }

    // stage P (16 x 64) to LDS so it can be re-read in the A layout
#pragma unroll
    for (int nt = 0; nt < 4; ++nt) {
      const int n = nt * 16 + ln;
#pragma unroll
      for (int r = 0; r < 8; ++r) {
        const int m = r + hh * 8;
        Pm[h][m * 72 + n] = (__bf16)sacc[nt][r];
      }
    }

    // out = P (16x64) @ V (64x32): 2 K chunks x 2 d tiles
    f32x8 oacc[2] = {zero8f(), zero8f()};
#pragma unroll
    for (int kc = 0; kc < 2; ++kc) {
      bf16x16 ap;
      const int o2 = kc * 32 + hh * 16;
#pragma unroll
      for (int i = 0; i < 8; ++i) {
        ap[i]     = Pm[h][ln * 72 + o2 + i];
        ap[i + 8] = Pm[h][ln * 72 + o2 + 8 + i];
      }
#pragma unroll
      for (int nt2 = 0; nt2 < 2; ++nt2) {
        bf16x16 bv = ld16bf(&vT[h][(nt2 * 16 + ln) * 72 + kc * 32 + hh * 16]);
        oacc[nt2] = wmma_bf16(ap, bv, oacc[nt2]);
      }
    }

    // store rows < 49
#pragma unroll
    for (int nt2 = 0; nt2 < 2; ++nt2) {
#pragma unroll
      for (int r = 0; r < 8; ++r) {
        const int m = mt * 16 + r + hh * 8;
        if (m < 49)
          attn_out[((size_t)s * 49 + m) * 96 + h * 32 + nt2 * 16 + ln] =
              (__bf16)oacc[nt2][r];
      }
    }
  }
}

// ---------------------------------------------------------------------------
// Kernel 3: temporal attention, L = T = 8. One thread per (seq, head, row).
// ---------------------------------------------------------------------------
__global__ void __launch_bounds__(256)
temporal_attn_kernel(const __bf16* __restrict__ qkv, __bf16* __restrict__ attn_out) {
  const int tid  = blockIdx.x * 256 + threadIdx.x;
  const int trow = tid & 7;
  const int h    = (tid >> 3) % 3;
  const int u    = tid / 24;                       // (b*49 + n)
  if (u >= B_ * N_) return;
  const int b = u / 49, n = u % 49;

  size_t rowbase[8];
#pragma unroll
  for (int t = 0; t < 8; ++t)
    rowbase[t] = ((size_t)(b * 8 + t) * 49 + n) * 288 + h * 32;

  float q[32];
#pragma unroll
  for (int d = 0; d < 32; ++d) q[d] = (float)qkv[rowbase[trow] + d] * SCALE;

  float sc[8];
  float mx = -1e30f;
#pragma unroll
  for (int t2 = 0; t2 < 8; ++t2) {
    float s = 0.0f;
#pragma unroll
    for (int d = 0; d < 32; ++d) s += q[d] * (float)qkv[rowbase[t2] + 96 + d];
    sc[t2] = s;
    mx = fmaxf(mx, s);
  }
  float ssum = 0.0f;
#pragma unroll
  for (int t2 = 0; t2 < 8; ++t2) { sc[t2] = __expf(sc[t2] - mx); ssum += sc[t2]; }
  const float inv = 1.0f / ssum;

  float o[32];
#pragma unroll
  for (int d = 0; d < 32; ++d) o[d] = 0.0f;
#pragma unroll
  for (int t2 = 0; t2 < 8; ++t2) {
    const float p = sc[t2] * inv;
#pragma unroll
    for (int d = 0; d < 32; ++d) o[d] += p * (float)qkv[rowbase[t2] + 192 + d];
  }
#pragma unroll
  for (int d = 0; d < 32; ++d)
    attn_out[((size_t)u * 8 + trow) * 96 + h * 32 + d] = (__bf16)o[d];
}

// ---------------------------------------------------------------------------
// Kernel 4: projection GEMM (TOK x 96) bf16 @ (96 x 96) f32 + bias -> f32 out.
// mode 0: identity row mapping (x_sp). mode 1: (b*49+n)*8+t -> (b*8+t)*49+n (x_t).
// ---------------------------------------------------------------------------
__global__ void __launch_bounds__(256)
proj_gemm_kernel(const __bf16* __restrict__ in, const float* __restrict__ w,
                 const float* __restrict__ bias, float* __restrict__ out, int mode) {
  __shared__ __bf16 wT[96 * 104];
  for (int i = threadIdx.x; i < 96 * 96; i += 256) {
    int n = i / 96, k = i % 96;
    wT[n * 104 + k] = (__bf16)w[(size_t)k * 96 + n];
  }
  __syncthreads();

  const int lane = threadIdx.x & 31;
  const int wv   = threadIdx.x >> 5;
  const int m0   = (blockIdx.x * 8 + wv) * 16;
  const int hh   = lane >> 4;
  const int ln   = lane & 15;

  bf16x16 a[3];
#pragma unroll
  for (int kc = 0; kc < 3; ++kc) {
    const __bf16* ar = in + (size_t)(m0 + ln) * 96 + kc * 32;
    const int o = hh * 16;
#pragma unroll
    for (int i = 0; i < 8; ++i) {
      a[kc][i]     = ar[o + i];
      a[kc][i + 8] = ar[o + 8 + i];
    }
  }

  for (int nt = 0; nt < 6; ++nt) {
    const float bv = bias[nt * 16 + ln];
    f32x8 c;
#pragma unroll
    for (int r = 0; r < 8; ++r) c[r] = bv;
#pragma unroll
    for (int kc = 0; kc < 3; ++kc) {
      bf16x16 bm = ld16bf(&wT[(nt * 16 + ln) * 104 + kc * 32 + hh * 16]);
      c = wmma_bf16(a[kc], bm, c);
    }
#pragma unroll
    for (int r = 0; r < 8; ++r) {
      const int m = m0 + r + hh * 8;
      size_t orow;
      if (mode == 0) {
        orow = (size_t)m;
      } else {
        const int b = m / 392, rem = m % 392;
        const int n = rem / 8, t = rem % 8;
        orow = (size_t)(b * 8 + t) * 49 + n;
      }
      out[orow * 96 + nt * 16 + ln] = c[r];
    }
  }
}

// ---------------------------------------------------------------------------
extern "C" void kernel_launch(void* const* d_in, const int* in_sizes, int n_in,
                              void* d_out, int out_size, void* d_ws, size_t ws_size,
                              hipStream_t stream) {
  (void)in_sizes; (void)n_in; (void)out_size; (void)ws_size;
  const float* x           = (const float*)d_in[0];
  const float* mask        = (const float*)d_in[1];
  const float* qkv_w       = (const float*)d_in[2];
  const float* qkv_b       = (const float*)d_in[3];
  const float* proj_sp_w   = (const float*)d_in[4];
  const float* proj_sp_b   = (const float*)d_in[5];
  const float* proj_temp_w = (const float*)d_in[6];
  const float* proj_temp_b = (const float*)d_in[7];
  float* out = (float*)d_out;

  __bf16* qkv     = (__bf16*)d_ws;                       // TOK*288 bf16
  __bf16* attn_sp = qkv + TOK * 288;                     // TOK*96 bf16
  __bf16* attn_t  = attn_sp + TOK * 96;                  // TOK*96 bf16

  const int mblocks = (int)(TOK / 16 / 8);               // 3136

  qkv_gemm_kernel<<<mblocks, 256, 0, stream>>>(x, qkv_w, qkv_b, qkv);
  spatial_attn_kernel<<<B_ * T_, 96, 0, stream>>>(qkv, mask, attn_sp);
  {
    const int tthreads = B_ * N_ * 3 * 8;
    temporal_attn_kernel<<<(tthreads + 255) / 256, 256, 0, stream>>>(qkv, attn_t);
  }
  // outputs: x_t first, then x_sp
  proj_gemm_kernel<<<mblocks, 256, 0, stream>>>(attn_sp, proj_sp_w, proj_sp_b,
                                                out + TOK * 96, 0);
  proj_gemm_kernel<<<mblocks, 256, 0, stream>>>(attn_t, proj_temp_w, proj_temp_b,
                                                out, 1);
}
